// MultiHeadAttention_81346680586718
// MI455X (gfx1250) — compile-verified, matrix operand order checked by reading, not run on hardware
//
#include <hip/hip_runtime.h>
#include <hip/hip_bf16.h>

typedef _Float16 h16;
typedef __attribute__((ext_vector_type(16))) _Float16 v16h;
typedef __attribute__((ext_vector_type(8)))  _Float16 v8h;
typedef __attribute__((ext_vector_type(4)))  _Float16 v4h;
typedef __attribute__((ext_vector_type(8)))  float    v8f;
typedef __attribute__((ext_vector_type(4)))  unsigned int v4u;
typedef __attribute__((ext_vector_type(8)))  unsigned int v8u;

static constexpr int Bb  = 2;
static constexpr int Ss  = 2048;
static constexpr int Dd  = 1024;
static constexpr int Hh  = 16;
static constexpr int HDd = 64;
static constexpr int Mrows = Bb * Ss;      // 4096

// ---------------------------------------------------------------------------
// Fragment loader: per-lane contiguous-in-K chunks, shared by A- and B-frags.
// lane<16 : row = lane,    K chunks {0..7, 16..23}
// lane>=16: row = lane-16, K chunks {8..15, 24..31}
// Works for global or LDS pointers (generic addressing).
// ---------------------------------------------------------------------------
__device__ __forceinline__ v16h load_frag(const h16* __restrict__ base, int ld, int lane) {
  const int hi  = lane >> 4;
  const int row = lane & 15;
  const h16* p = base + (size_t)row * ld + hi * 8;
  v8h lo = *(const v8h*)p;
  v8h hh = *(const v8h*)(p + 16);
  v16h r;
#pragma unroll
  for (int i = 0; i < 8; ++i) { r[i] = lo[i]; r[i + 8] = hh[i]; }
  return r;
}

__device__ __forceinline__ v8f wmma_f16(v16h a, v16h b, v8f c) {
  return __builtin_amdgcn_wmma_f32_16x16x32_f16(false, a, false, b, (short)0, c, false, false);
}

// ---------------------------------------------------------------------------
// Tensor Data Mover: 2-D tile (rows x cols, f16) from global -> LDS.
// D# per CDNA5 ISA §8: group0 = {count|flags, lds_addr, global_addr, type=2},
// group1 = {mask/data_size, tensor dims, tile dims, dim0 stride}.
// Issued via inline asm (2-operand form: VADDR2/3 = NULL -> up-to-2D tensor).
// ---------------------------------------------------------------------------
__device__ __forceinline__ void tdm_load_2d(unsigned lds_off, const void* gptr,
                                            unsigned rows, unsigned cols,
                                            unsigned stride_elems) {
  const unsigned long long ga = (unsigned long long)(size_t)gptr;
  v4u g0;
  g0[0] = 1u;                                                  // count=1, user D#
  g0[1] = lds_off;                                             // lds_addr (bytes)
  g0[2] = (unsigned)ga;                                        // global_addr[31:0]
  g0[3] = (unsigned)((ga >> 32) & 0x01FFFFFFu) | 0x80000000u;  // addr[56:32] | type=2
  v8u g1;
  g1[0] = 0x00010000u;        // workgroup_mask=0, data_size=1 (2 bytes)
  g1[1] = cols << 16;         // tensor_dim0[15:0]  (bits 63:48)
  g1[2] = rows << 16;         // tensor_dim0 hi=0, tensor_dim1[15:0] (bits 95:80)
  g1[3] = cols << 16;         // tensor_dim1 hi=0, tile_dim0 (bits 127:112)
  g1[4] = rows;               // tile_dim1 (bits 143:128), tile_dim2=0
  g1[5] = stride_elems;       // tensor_dim0_stride[31:0]
  g1[6] = 0u;                 // stride hi, tensor_dim1_stride lo (unused for 2D tile)
  g1[7] = 0u;
  asm volatile("tensor_load_to_lds %0, %1" :: "s"(g0), "s"(g1) : "memory");
}

// ---------------------------------------------------------------------------
// fp32 -> fp16 conversion, 4 elements/thread
// ---------------------------------------------------------------------------
__global__ void __launch_bounds__(256) cvt_f32_f16(const float* __restrict__ in,
                                                   h16* __restrict__ out, int n) {
  int i = (blockIdx.x * 256 + threadIdx.x) * 4;
  if (i < n) {
    float4 f = *(const float4*)(in + i);
    v4h o;
    o[0] = (h16)f.x; o[1] = (h16)f.y; o[2] = (h16)f.z; o[3] = (h16)f.w;
    *(v4h*)(out + i) = o;
  }
}

// ---------------------------------------------------------------------------
// GEMM: C[M=4096][N=1024] = A[M][1024] * W^T + bias   (W is [N][K], row-major)
// MODE 0: write f16 Q/K layout  [b][h][s][d]
// MODE 1: write f16 Vt layout   [b][h][d][s]
// MODE 2: write f32 row-major   [m][n]  (final output)
// One wave computes a 16x64 strip (A-frag reused across 4 WMMAs per k-step).
// ---------------------------------------------------------------------------
template <int MODE>
__global__ void __launch_bounds__(256) gemm1024(const h16* __restrict__ A,
                                                const h16* __restrict__ W,
                                                const float* __restrict__ bias,
                                                void* __restrict__ outp) {
  const int lane = threadIdx.x & 31;
  const int wave = threadIdx.x >> 5;
  const int gid  = blockIdx.x * 8 + wave;
  const int NT   = Dd / 64;              // 16 tiles along N
  const int m0   = (gid / NT) * 16;
  const int n0   = (gid % NT) * 64;

  v8f acc[4] = {};
  const h16* Abase = A + (size_t)m0 * Dd;

#pragma unroll 1
  for (int k = 0; k < Dd; k += 32) {
    __builtin_prefetch(Abase + k + 64, 0, 0);          // global_prefetch_b8
    v16h a = load_frag(Abase + k, Dd, lane);
#pragma unroll
    for (int j = 0; j < 4; ++j) {
      v16h b = load_frag(W + (size_t)(n0 + j * 16) * Dd + k, Dd, lane);
      acc[j] = wmma_f16(a, b, acc[j]);
    }
  }

  const int hi = lane >> 4, col = lane & 15;
#pragma unroll
  for (int j = 0; j < 4; ++j) {
    const int n = n0 + j * 16 + col;
    const float bn = bias[n];
#pragma unroll
    for (int r = 0; r < 8; ++r) {
      const int m = m0 + r + hi * 8;
      const float v = acc[j][r] + bn;
      if (MODE == 2) {
        ((float*)outp)[(size_t)m * Dd + n] = v;
      } else {
        const int b = m / Ss, s = m % Ss;
        const int h = n / HDd, d = n % HDd;
        const size_t off = (MODE == 0)
            ? ((size_t)((b * Hh + h) * Ss + s)) * HDd + d
            : ((size_t)((b * Hh + h) * HDd + d)) * Ss + s;
        ((h16*)outp)[off] = (h16)v;
      }
    }
  }
}

// ---------------------------------------------------------------------------
// Flash attention: one wave = 16 queries x HD=64.  Key tiles of 64, staged
// into LDS by the Tensor Data Mover (double-buffered on TENSORcnt); all 4
// waves of the block share the K/V tiles from LDS.
// Q/K [b][h][s][d], Vt [b][h][d][s] (f16).  O f16 in [b][s][h*64+d].
// ---------------------------------------------------------------------------
__global__ void __launch_bounds__(128) flash_attn(const h16* __restrict__ Q,
                                                  const h16* __restrict__ K,
                                                  const h16* __restrict__ Vt,
                                                  const int* __restrict__ mask,
                                                  h16* __restrict__ O) {
  __shared__ h16 Kbuf[2][64 * 64];     // 16 KB : keys x d
  __shared__ h16 Vbuf[2][64 * 64];     // 16 KB : d x keys
  __shared__ h16 Plds[4][16 * 72];     // 9 KB  : per-wave P tile, 144B row stride

  const int lane = threadIdx.x & 31;
  const int wave = threadIdx.x >> 5;
  const int QT   = Ss / 64;
  int bid = blockIdx.x;
  const int qt = bid % QT;  bid /= QT;
  const int h  = bid % Hh;
  const int b  = bid / Hh;
  const int q0 = qt * 64 + wave * 16;

  const size_t head = (size_t)(b * Hh + h) * Ss * HDd;
  const h16* Qp = Q + head;
  const h16* Kp = K + head;
  const h16* Vp = Vt + (size_t)(b * Hh + h) * HDd * Ss;
  const int* mp = mask + b * Ss;

  const v16h qa0 = load_frag(Qp + (size_t)q0 * HDd +  0, HDd, lane);
  const v16h qa1 = load_frag(Qp + (size_t)q0 * HDd + 32, HDd, lane);

  v8f o0 = {}, o1 = {}, o2 = {}, o3 = {};
  float mrun[8], lrun[8];
#pragma unroll
  for (int r = 0; r < 8; ++r) { mrun[r] = -INFINITY; lrun[r] = 0.f; }

  h16* pl = Plds[wave];
  const int hi = lane >> 4, col = lane & 15;
  const float scale = 0.125f;            // 1/sqrt(64)

  // preload tile 0 (K: rows=64 keys x cols=64 d; V: rows=64 d x cols=64 keys)
  if (wave == 0) {
    tdm_load_2d((unsigned)(size_t)&Kbuf[0][0], Kp, 64, 64, HDd);
    tdm_load_2d((unsigned)(size_t)&Vbuf[0][0], Vp, 64, 64, Ss);
  }

  const int NT = Ss / 64;
  for (int it = 0; it < NT; ++it) {
    const int kt = it * 64;
    if (wave == 0) {
      if (it + 1 < NT) {
        tdm_load_2d((unsigned)(size_t)&Kbuf[(it + 1) & 1][0],
                    Kp + (size_t)(kt + 64) * HDd, 64, 64, HDd);
        tdm_load_2d((unsigned)(size_t)&Vbuf[(it + 1) & 1][0],
                    Vp + (kt + 64), 64, 64, Ss);
        asm volatile("s_wait_tensorcnt 0x2" ::: "memory");  // current pair done
      } else {
        asm volatile("s_wait_tensorcnt 0x0" ::: "memory");
      }
    }
    __syncthreads();                     // tile visible to all waves
    const h16* Kt = &Kbuf[it & 1][0];
    const h16* Vv = &Vbuf[it & 1][0];

    // --- scores: 4 groups of 16 keys, QK^T over d=64 (2 chained WMMAs each) ---
    v8f sg[4];
#pragma unroll
    for (int g = 0; g < 4; ++g) {
      v16h bk0 = load_frag(Kt + (g * 16) * 64 +  0, 64, lane);
      v16h bk1 = load_frag(Kt + (g * 16) * 64 + 32, 64, lane);
      v8f z = {};
      z = wmma_f16(qa0, bk0, z);
      sg[g] = wmma_f16(qa1, bk1, z);
    }

    // --- mask + scale (column == key index, fixed per lane per group) ---
    bool mz[4];
#pragma unroll
    for (int g = 0; g < 4; ++g) mz[g] = (mp[kt + g * 16 + col] == 0);
#pragma unroll
    for (int g = 0; g < 4; ++g)
#pragma unroll
      for (int r = 0; r < 8; ++r)
        sg[g][r] = mz[g] ? -1.0e9f : sg[g][r] * scale;

    // --- one online-softmax update per 64 keys ---
#pragma unroll
    for (int r = 0; r < 8; ++r) {
      float t = fmaxf(fmaxf(sg[0][r], sg[1][r]), fmaxf(sg[2][r], sg[3][r]));
#pragma unroll
      for (int off = 1; off < 16; off <<= 1) t = fmaxf(t, __shfl_xor(t, off, 32));
      const float mnew = fmaxf(mrun[r], t);
      const float corr = __expf(mrun[r] - mnew);
      mrun[r] = mnew;
      o0[r] *= corr; o1[r] *= corr; o2[r] *= corr; o3[r] *= corr;
      float rs = 0.f;
#pragma unroll
      for (int g = 0; g < 4; ++g) {
        const float p = __expf(sg[g][r] - mnew);
        rs += p;
        pl[(r + hi * 8) * 72 + g * 16 + col] = (h16)p;
      }
#pragma unroll
      for (int off = 1; off < 16; off <<= 1) rs += __shfl_xor(rs, off, 32);
      lrun[r] = lrun[r] * corr + rs;
    }
    asm volatile("s_wait_dscnt 0x0" ::: "memory");   // wave-local P RAW

    // --- repack P (16x64) as two A-fragments from LDS ---
    const h16* pr = pl + (size_t)col * 72 + hi * 8;
    v8h c0 = *(const v8h*)(pr +  0);
    v8h c1 = *(const v8h*)(pr + 16);
    v8h c2 = *(const v8h*)(pr + 32);
    v8h c3 = *(const v8h*)(pr + 48);
    v16h pa0, pa1;
#pragma unroll
    for (int i = 0; i < 8; ++i) {
      pa0[i] = c0[i]; pa0[i + 8] = c1[i];
      pa1[i] = c2[i]; pa1[i + 8] = c3[i];
    }

    // --- O += P * V  (V rows are d, contiguous along keys in LDS) ---
    {
      v16h vb0, vb1;
      vb0 = load_frag(Vv + (0 * 16) * 64 + 0, 64, lane);
      vb1 = load_frag(Vv + (0 * 16) * 64 + 32, 64, lane);
      o0 = wmma_f16(pa1, vb1, wmma_f16(pa0, vb0, o0));
      vb0 = load_frag(Vv + (1 * 16) * 64 + 0, 64, lane);
      vb1 = load_frag(Vv + (1 * 16) * 64 + 32, 64, lane);
      o1 = wmma_f16(pa1, vb1, wmma_f16(pa0, vb0, o1));
      vb0 = load_frag(Vv + (2 * 16) * 64 + 0, 64, lane);
      vb1 = load_frag(Vv + (2 * 16) * 64 + 32, 64, lane);
      o2 = wmma_f16(pa1, vb1, wmma_f16(pa0, vb0, o2));
      vb0 = load_frag(Vv + (3 * 16) * 64 + 0, 64, lane);
      vb1 = load_frag(Vv + (3 * 16) * 64 + 32, 64, lane);
      o3 = wmma_f16(pa1, vb1, wmma_f16(pa0, vb0, o3));
    }
    __syncthreads();                     // everyone done with this buffer
  }

  // --- normalize and store O in [b][s][h*64+d] f16 layout ---
#pragma unroll
  for (int r = 0; r < 8; ++r) {
    const float inv = 1.0f / lrun[r];
    const int q = q0 + r + hi * 8;
    h16* ob = O + ((size_t)b * Ss + q) * Dd + h * HDd;
    ob[0 * 16 + col] = (h16)(o0[r] * inv);
    ob[1 * 16 + col] = (h16)(o1[r] * inv);
    ob[2 * 16 + col] = (h16)(o2[r] * inv);
    ob[3 * 16 + col] = (h16)(o3[r] * inv);
  }
}

// ---------------------------------------------------------------------------
extern "C" void kernel_launch(void* const* d_in, const int* in_sizes, int n_in,
                              void* d_out, int out_size, void* d_ws, size_t ws_size,
                              hipStream_t stream) {
  const float* xq = (const float*)d_in[0];
  const float* xk = (const float*)d_in[1];
  const float* xv = (const float*)d_in[2];
  const int*   mk = (const int*)  d_in[3];
  const float* wq = (const float*)d_in[4];
  const float* bq = (const float*)d_in[5];
  const float* wk = (const float*)d_in[6];
  const float* bk = (const float*)d_in[7];
  const float* wv = (const float*)d_in[8];
  const float* bv = (const float*)d_in[9];
  const float* wo = (const float*)d_in[10];
  const float* bo = (const float*)d_in[11];

  h16* ws = (h16*)d_ws;
  const size_t XN = (size_t)Mrows * Dd;   // 4,194,304 halves
  const size_t WN = (size_t)Dd * Dd;      // 1,048,576 halves
  h16* Xq   = ws;
  h16* Xk   = Xq + XN;
  h16* Xv   = Xk + XN;
  h16* Wq16 = Xv + XN;
  h16* Wk16 = Wq16 + WN;
  h16* Wv16 = Wk16 + WN;
  h16* Wo16 = Wv16 + WN;
  h16* Q16  = Wo16 + WN;
  h16* K16  = Q16 + XN;
  h16* Vt16 = K16 + XN;
  h16* O16  = Xq;   // Xq is dead after the Q projection -> safe reuse

  // fp32 -> fp16 staging
  cvt_f32_f16<<<(int)(XN / 1024), 256, 0, stream>>>(xq, Xq, (int)XN);
  cvt_f32_f16<<<(int)(XN / 1024), 256, 0, stream>>>(xk, Xk, (int)XN);
  cvt_f32_f16<<<(int)(XN / 1024), 256, 0, stream>>>(xv, Xv, (int)XN);
  cvt_f32_f16<<<(int)(WN / 1024), 256, 0, stream>>>(wq, Wq16, (int)WN);
  cvt_f32_f16<<<(int)(WN / 1024), 256, 0, stream>>>(wk, Wk16, (int)WN);
  cvt_f32_f16<<<(int)(WN / 1024), 256, 0, stream>>>(wv, Wv16, (int)WN);
  cvt_f32_f16<<<(int)(WN / 1024), 256, 0, stream>>>(wo, Wo16, (int)WN);

  // projections (512 blocks x 8 waves = 4096 wave-tiles of 16x64)
  gemm1024<0><<<512, 256, 0, stream>>>(Xq, Wq16, bq, Q16);
  gemm1024<0><<<512, 256, 0, stream>>>(Xk, Wk16, bk, K16);
  gemm1024<1><<<512, 256, 0, stream>>>(Xv, Wv16, bv, Vt16);

  // flash attention: B*H*(S/64) = 1024 blocks x 4 waves
  flash_attn<<<Bb * Hh * (Ss / 64), 128, 0, stream>>>(Q16, K16, Vt16, mk, O16);

  // output projection -> fp32 d_out
  gemm1024<2><<<512, 256, 0, stream>>>(O16, Wo16, bo, d_out);
}
